// GatedScan_42606075576728
// MI455X (gfx1250) — matrix operand order repaired and checked by reading, not compiled, not run on hardware
//
#include <hip/hip_runtime.h>
#include <hip/hip_bf16.h>

typedef __attribute__((ext_vector_type(16))) __bf16 v16bf;
typedef __attribute__((ext_vector_type(8)))  __bf16 v8bf;
typedef __attribute__((ext_vector_type(8)))  float  v8f;

namespace cfg {
constexpr int B  = 4;
constexpr int T  = 8192;
constexpr int D  = 1024;
constexpr int S  = 1024;
constexpr int K  = 32;
constexpr int NC = T / K;      // 256 chunks
constexpr int M  = B * T;      // 32768 GEMM rows
}

// ---------------------------------------------------------------------------
// WMMA operand loaders: layouts per CDNA5 ISA §7.12.2
// A (16x32 bf16, MxK): lane lo=l&15 -> row M=lo; hi=l>>4.
//   element e in [0,16): K = (e>>3)*16 + hi*8 + (e&7)   -> two 16B loads
// B (32x16 bf16, KxN): lane lo -> col N=lo; element e: K = (e>>3)*16 + hi*8 + (e&7)
//   -> same K order as A -> two 16B loads
// ---------------------------------------------------------------------------
__device__ __forceinline__ v16bf load_tileA(const __bf16* base, int ld, int lane) {
  const int lo = lane & 15, hi = lane >> 4;
  const __bf16* p = base + (size_t)lo * ld + hi * 8;
  v8bf x0 = *(const v8bf*)(p);
  v8bf x1 = *(const v8bf*)(p + 16);
  v16bf r;
#pragma unroll
  for (int i = 0; i < 8; ++i) { r[i] = x0[i]; r[i + 8] = x1[i]; }
  return r;
}

__device__ __forceinline__ v16bf load_tileB(const __bf16* base, int ld, int lane) {
  const int lo = lane & 15, hi = lane >> 4;
  const __bf16* p = base + (size_t)lo * ld + hi * 8;
  v8bf x0 = *(const v8bf*)(p);
  v8bf x1 = *(const v8bf*)(p + 16);
  v16bf r;
#pragma unroll
  for (int i = 0; i < 8; ++i) { r[i] = x0[i]; r[i + 8] = x1[i]; }
  return r;
}

__device__ __forceinline__ v8f wmma_bf16(v16bf a, v16bf b, v8f c) {
  return __builtin_amdgcn_wmma_f32_16x16x32_bf16(false, a, false, b,
                                                 (short)0, c, false, false);
}

// ---------------------------------------------------------------------------
// Kernel 0: f32 -> bf16 conversion (x, Wg, Wi, Wo)
// ---------------------------------------------------------------------------
__global__ void cvt_f32_bf16(const float* __restrict__ src,
                             __bf16* __restrict__ dst, int n) {
  int i = blockIdx.x * blockDim.x + threadIdx.x;
  const int stride = gridDim.x * blockDim.x;
  for (; i < n; i += stride) dst[i] = (__bf16)src[i];
}

// ---------------------------------------------------------------------------
// Kernel 1: dual projection GEMM (gates + drive), fused sigmoid epilogue.
// Manual double-buffer, k-step 64: set A is loop-carried, set B intra-
// iteration; no cur<-next copies exist in the source at all.
// Per wave: 32(M) x 32(N) tile, two accumulator sets (Wg and Wi paths).
// ---------------------------------------------------------------------------
__global__ void __launch_bounds__(128)
proj_gemm(const __bf16* __restrict__ xbf,
          const __bf16* __restrict__ Wgbf, const __bf16* __restrict__ Wibf,
          const float* __restrict__ bg, const float* __restrict__ bi,
          float2* __restrict__ gd) {
  using namespace cfg;
  const int lane = threadIdx.x & 31;
  const int wave = threadIdx.x >> 5;
  const int n0 = blockIdx.x * 32;
  const int m0 = blockIdx.y * 128 + wave * 32;

  const __bf16* pa0 = xbf  + (size_t)m0 * D;
  const __bf16* pa1 = xbf  + (size_t)(m0 + 16) * D;
  const __bf16* pg0 = Wgbf + (size_t)n0 * D;
  const __bf16* pg1 = Wgbf + (size_t)(n0 + 16) * D;
  const __bf16* pi0 = Wibf + (size_t)n0 * D;
  const __bf16* pi1 = Wibf + (size_t)(n0 + 16) * D;

  const v8f zero = {0.f, 0.f, 0.f, 0.f, 0.f, 0.f, 0.f, 0.f};
  v8f cg[2][2], ci[2][2];
#pragma unroll
  for (int mi = 0; mi < 2; ++mi)
#pragma unroll
    for (int ni = 0; ni < 2; ++ni) { cg[mi][ni] = zero; ci[mi][ni] = zero; }

  // Buffer set A: k = 0
  v16bf a0A = load_tileA(pa0, D, lane);
  v16bf a1A = load_tileA(pa1, D, lane);
  v16bf g0A = load_tileB(pg0, D, lane);
  v16bf g1A = load_tileB(pg1, D, lane);
  v16bf i0A = load_tileB(pi0, D, lane);
  v16bf i1A = load_tileB(pi1, D, lane);

  for (int k0 = 0; k0 < D; k0 += 64) {
    // ---- stage 0: load k0+32 into set B (always in range), compute set A
    const int kb = k0 + 32;
    __builtin_prefetch(pa0 + kb + 128, 0, 3);   // WGP-scope prefetch
    const v16bf a0B = load_tileA(pa0 + kb, D, lane);
    const v16bf a1B = load_tileA(pa1 + kb, D, lane);
    const v16bf g0B = load_tileB(pg0 + kb, D, lane);
    const v16bf g1B = load_tileB(pg1 + kb, D, lane);
    const v16bf i0B = load_tileB(pi0 + kb, D, lane);
    const v16bf i1B = load_tileB(pi1 + kb, D, lane);

    cg[0][0] = wmma_bf16(a0A, g0A, cg[0][0]);
    cg[0][1] = wmma_bf16(a0A, g1A, cg[0][1]);
    cg[1][0] = wmma_bf16(a1A, g0A, cg[1][0]);
    cg[1][1] = wmma_bf16(a1A, g1A, cg[1][1]);
    ci[0][0] = wmma_bf16(a0A, i0A, ci[0][0]);
    ci[0][1] = wmma_bf16(a0A, i1A, ci[0][1]);
    ci[1][0] = wmma_bf16(a1A, i0A, ci[1][0]);
    ci[1][1] = wmma_bf16(a1A, i1A, ci[1][1]);

    // ---- stage 1: load k0+64 into set A (guarded), compute set B
    const int ka = k0 + 64;
    if (ka < D) {
      __builtin_prefetch(pa0 + ka + 128, 0, 3);
      a0A = load_tileA(pa0 + ka, D, lane);
      a1A = load_tileA(pa1 + ka, D, lane);
      g0A = load_tileB(pg0 + ka, D, lane);
      g1A = load_tileB(pg1 + ka, D, lane);
      i0A = load_tileB(pi0 + ka, D, lane);
      i1A = load_tileB(pi1 + ka, D, lane);
    }

    cg[0][0] = wmma_bf16(a0B, g0B, cg[0][0]);
    cg[0][1] = wmma_bf16(a0B, g1B, cg[0][1]);
    cg[1][0] = wmma_bf16(a1B, g0B, cg[1][0]);
    cg[1][1] = wmma_bf16(a1B, g1B, cg[1][1]);
    ci[0][0] = wmma_bf16(a0B, i0B, ci[0][0]);
    ci[0][1] = wmma_bf16(a0B, i1B, ci[0][1]);
    ci[1][0] = wmma_bf16(a1B, i0B, ci[1][0]);
    ci[1][1] = wmma_bf16(a1B, i1B, ci[1][1]);
  }

  // C layout: VGPR j -> row j (lanes 0-15) / row j+8 (lanes 16-31), col = lane&15
  const int hi = lane >> 4, lo = lane & 15;
#pragma unroll
  for (int mi = 0; mi < 2; ++mi)
#pragma unroll
    for (int ni = 0; ni < 2; ++ni)
#pragma unroll
      for (int j = 0; j < 8; ++j) {
        const int row = m0 + mi * 16 + hi * 8 + j;
        const int col = n0 + ni * 16 + lo;
        const float tg = cg[mi][ni][j] + bg[col];
        const float g  = 1.f / (1.f + __expf(-tg));
        const float dv = (1.f - g) * (ci[mi][ni][j] + bi[col]);
        gd[(size_t)row * S + col] = make_float2(g, dv);   // one b64 store
      }
}

// ---------------------------------------------------------------------------
// Scan pass 1: per-chunk affine aggregates  h_end = P*h_start + Q
// One thread per (b, chunk, s); fully parallel (1M threads).
// ---------------------------------------------------------------------------
__global__ void chunk_agg(const float2* __restrict__ gd,
                          float* __restrict__ P, float* __restrict__ Q) {
  using namespace cfg;
  const int s  = blockIdx.x * blockDim.x + threadIdx.x;
  const int bc = blockIdx.y;            // b*NC + c
  const int b  = bc / NC, c = bc % NC;
  const size_t base = ((size_t)b * T + (size_t)c * K) * S + s;
  float p = 1.f, q = 0.f;
#pragma unroll 4
  for (int t = 0; t < K; ++t) {
    const float2 v = gd[base + (size_t)t * S];
    const float at = fmaxf(v.x, 1e-6f);
    p *= at;
    q = at * q + v.y;
  }
  P[(size_t)bc * S + s] = p;
  Q[(size_t)bc * S + s] = q;
}

// ---------------------------------------------------------------------------
// Scan pass 2: sequential scan over the 256 chunk aggregates per (b,s).
// ---------------------------------------------------------------------------
__global__ void chunk_scan(const float* __restrict__ P, const float* __restrict__ Q,
                           float* __restrict__ H) {
  using namespace cfg;
  const int s = blockIdx.x * blockDim.x + threadIdx.x;
  const int b = blockIdx.y;
  float h = 0.f;
  for (int c = 0; c < NC; ++c) {
    const size_t idx = ((size_t)b * NC + c) * S + s;
    H[idx] = h;
    h = P[idx] * h + Q[idx];
  }
}

// ---------------------------------------------------------------------------
// Scan pass 3: replay each chunk from its start state; emit states as bf16
// (feeds the WMMA output GEMM directly).
// ---------------------------------------------------------------------------
__global__ void chunk_apply(const float2* __restrict__ gd,
                            const float* __restrict__ H, __bf16* __restrict__ st) {
  using namespace cfg;
  const int s  = blockIdx.x * blockDim.x + threadIdx.x;
  const int bc = blockIdx.y;
  const int b  = bc / NC, c = bc % NC;
  const size_t base = ((size_t)b * T + (size_t)c * K) * S + s;
  float h = H[(size_t)bc * S + s];
#pragma unroll 4
  for (int t = 0; t < K; ++t) {
    const float2 v = gd[base + (size_t)t * S];
    h = fmaxf(v.x, 1e-6f) * h + v.y;
    st[base + (size_t)t * S] = (__bf16)h;
  }
}

// ---------------------------------------------------------------------------
// Kernel 3: output GEMM + residual + bias + LayerNorm, fully fused.
// Workgroup = 512 threads = 16 waves; owns 16 rows x full D=1024 columns.
// Wave w computes a 16x64 slab (4 WMMA N-tiles, manual double-buffer) into
// 64KB LDS; then wave w LayerNorms row w with wave32 shuffle reductions.
// ---------------------------------------------------------------------------
__global__ void __launch_bounds__(512)
out_gemm_ln(const __bf16* __restrict__ stbf, const __bf16* __restrict__ Wobf,
            const float* __restrict__ x, const float* __restrict__ bo,
            const float* __restrict__ lng, const float* __restrict__ lnb,
            float* __restrict__ out) {
  using namespace cfg;
  __shared__ float ytile[16][1024];     // 64 KB of the 320 KB WGP LDS

  const int lane = threadIdx.x & 31;
  const int wave = threadIdx.x >> 5;
  const int m0 = blockIdx.x * 16;
  const int n0 = wave * 64;

  const __bf16* pa  = stbf + (size_t)m0 * S;
  const __bf16* pb0 = Wobf + (size_t)(n0 +  0) * S;
  const __bf16* pb1 = Wobf + (size_t)(n0 + 16) * S;
  const __bf16* pb2 = Wobf + (size_t)(n0 + 32) * S;
  const __bf16* pb3 = Wobf + (size_t)(n0 + 48) * S;

  const v8f zero = {0.f, 0.f, 0.f, 0.f, 0.f, 0.f, 0.f, 0.f};
  v8f acc[4];
#pragma unroll
  for (int ni = 0; ni < 4; ++ni) acc[ni] = zero;

  // Buffer set A: k = 0
  v16bf aA  = load_tileA(pa,  S, lane);
  v16bf b0A = load_tileB(pb0, S, lane);
  v16bf b1A = load_tileB(pb1, S, lane);
  v16bf b2A = load_tileB(pb2, S, lane);
  v16bf b3A = load_tileB(pb3, S, lane);

  for (int k0 = 0; k0 < S; k0 += 64) {
    // ---- stage 0: load k0+32 into set B, compute set A
    const int kb = k0 + 32;
    __builtin_prefetch(pa + kb + 128, 0, 3);
    const v16bf aB  = load_tileA(pa  + kb, S, lane);
    const v16bf b0B = load_tileB(pb0 + kb, S, lane);
    const v16bf b1B = load_tileB(pb1 + kb, S, lane);
    const v16bf b2B = load_tileB(pb2 + kb, S, lane);
    const v16bf b3B = load_tileB(pb3 + kb, S, lane);

    acc[0] = wmma_bf16(aA, b0A, acc[0]);
    acc[1] = wmma_bf16(aA, b1A, acc[1]);
    acc[2] = wmma_bf16(aA, b2A, acc[2]);
    acc[3] = wmma_bf16(aA, b3A, acc[3]);

    // ---- stage 1: load k0+64 into set A (guarded), compute set B
    const int ka = k0 + 64;
    if (ka < S) {
      __builtin_prefetch(pa + ka + 128, 0, 3);
      aA  = load_tileA(pa  + ka, S, lane);
      b0A = load_tileB(pb0 + ka, S, lane);
      b1A = load_tileB(pb1 + ka, S, lane);
      b2A = load_tileB(pb2 + ka, S, lane);
      b3A = load_tileB(pb3 + ka, S, lane);
    }

    acc[0] = wmma_bf16(aB, b0B, acc[0]);
    acc[1] = wmma_bf16(aB, b1B, acc[1]);
    acc[2] = wmma_bf16(aB, b2B, acc[2]);
    acc[3] = wmma_bf16(aB, b3B, acc[3]);
  }

  // Scatter accumulators into the shared row-panel.
  const int hi = lane >> 4, lo = lane & 15;
#pragma unroll
  for (int ni = 0; ni < 4; ++ni)
#pragma unroll
    for (int j = 0; j < 8; ++j)
      ytile[hi * 8 + j][n0 + ni * 16 + lo] = acc[ni][j];
  __syncthreads();

  // Wave w: LayerNorm of row (m0 + w). Pass 1: residual + bias + moments.
  const int m = m0 + wave;
  float ssum = 0.f, ssq = 0.f;
#pragma unroll 4
  for (int i = 0; i < 32; ++i) {
    const int idx = lane + 32 * i;
    const float y = ytile[wave][idx] + x[(size_t)m * D + idx] + bo[idx];
    ytile[wave][idx] = y;
    ssum += y;
    ssq  += y * y;
  }
#pragma unroll
  for (int off = 16; off > 0; off >>= 1) {   // wave32 butterfly
    ssum += __shfl_xor(ssum, off, 32);
    ssq  += __shfl_xor(ssq,  off, 32);
  }
  const float mu   = ssum * (1.0f / (float)D);
  const float var  = ssq * (1.0f / (float)D) - mu * mu;
  const float rstd = rsqrtf(var + 1e-5f);

#pragma unroll 4
  for (int i = 0; i < 32; ++i) {
    const int idx = lane + 32 * i;
    const float y = ytile[wave][idx];
    out[(size_t)m * D + idx] = (y - mu) * rstd * lng[idx] + lnb[idx];
  }
}

// ---------------------------------------------------------------------------
// Host-side launch sequence
// ---------------------------------------------------------------------------
extern "C" void kernel_launch(void* const* d_in, const int* in_sizes, int n_in,
                              void* d_out, int out_size, void* d_ws, size_t ws_size,
                              hipStream_t stream) {
  using namespace cfg;
  const float* x   = (const float*)d_in[0];
  const float* Wg  = (const float*)d_in[1];
  const float* bg  = (const float*)d_in[2];
  const float* Wi  = (const float*)d_in[3];
  const float* bi  = (const float*)d_in[4];
  const float* Wo  = (const float*)d_in[5];
  const float* bo  = (const float*)d_in[6];
  const float* lng = (const float*)d_in[7];
  const float* lnb = (const float*)d_in[8];
  float* out = (float*)d_out;

  char* ws = (char*)d_ws;
  auto alloc = [&](size_t bytes) {
    char* p = ws;
    ws += (bytes + 255) & ~(size_t)255;
    return p;
  };
  __bf16* xbf   = (__bf16*)alloc((size_t)M * D * sizeof(__bf16));    //  64 MB
  __bf16* Wgbf  = (__bf16*)alloc((size_t)S * D * sizeof(__bf16));    //   2 MB
  __bf16* Wibf  = (__bf16*)alloc((size_t)S * D * sizeof(__bf16));    //   2 MB
  __bf16* Wobf  = (__bf16*)alloc((size_t)D * S * sizeof(__bf16));    //   2 MB
  float2* gd    = (float2*)alloc((size_t)M * S * sizeof(float2));    // 256 MB
  __bf16* stbf  = (__bf16*)alloc((size_t)M * S * sizeof(__bf16));    //  64 MB
  float*  aggP  = (float*)alloc((size_t)B * NC * S * sizeof(float)); //   4 MB
  float*  aggQ  = (float*)alloc((size_t)B * NC * S * sizeof(float)); //   4 MB
  float*  hst   = (float*)alloc((size_t)B * NC * S * sizeof(float)); //   4 MB

  // 0) precision conversion for the tensor-core path
  cvt_f32_bf16<<<2048, 256, 0, stream>>>(x,  xbf,  M * D);
  cvt_f32_bf16<<<512,  256, 0, stream>>>(Wg, Wgbf, S * D);
  cvt_f32_bf16<<<512,  256, 0, stream>>>(Wi, Wibf, S * D);
  cvt_f32_bf16<<<512,  256, 0, stream>>>(Wo, Wobf, D * S);

  // 1) fused dual projection GEMM (sigmoid + drive epilogue, packed float2)
  proj_gemm<<<dim3(S / 32, M / 128), 128, 0, stream>>>(
      xbf, Wgbf, Wibf, bg, bi, gd);

  // 2) chunked parallel scan: aggregate -> tiny scan -> replay
  chunk_agg<<<dim3(S / 256, B * NC), 256, 0, stream>>>(gd, aggP, aggQ);
  chunk_scan<<<dim3(S / 256, B),     256, 0, stream>>>(aggP, aggQ, hst);
  chunk_apply<<<dim3(S / 256, B * NC), 256, 0, stream>>>(gd, hst, stbf);

  // 3) output GEMM + residual + LayerNorm, fused
  out_gemm_ln<<<M / 16, 512, 0, stream>>>(stbf, Wobf, x, bo, lng, lnb, out);
}
